// CompressedGLAHead_62302795596591
// MI455X (gfx1250) — compile-verified, matrix operand order checked by reading, not run on hardware
//
#include <hip/hip_runtime.h>
#include <stdint.h>

// ---------------------------------------------------------------------------
// CompressedGLAHead on MI455X (gfx1250): bf16 WMMA everywhere parallel,
// j-sliced register-resident WMMA scan for the sequential part.
// B=4, S=2048, DM=1024, DK=64, DV=64, DC=128
// ---------------------------------------------------------------------------

#define B_    4
#define S_    2048
#define DM_   1024
#define DK_   64
#define DV_   64
#define DC_   128
#define NTOK  (B_ * S_)        // 8192
#define KV_   (DK_ * DV_)      // 4096

typedef __attribute__((ext_vector_type(16))) __bf16       v16bf;
typedef __attribute__((ext_vector_type(8)))  float        v8f;
typedef __attribute__((ext_vector_type(8)))  unsigned int u32x8;
typedef __attribute__((ext_vector_type(4)))  unsigned int u32x4;
typedef __attribute__((ext_vector_type(8)))  int          i32x8;
typedef __attribute__((ext_vector_type(4)))  int          i32x4;

#if defined(__has_builtin)
#if __has_builtin(__builtin_amdgcn_tensor_load_to_lds) && __has_builtin(__builtin_amdgcn_s_wait_tensorcnt)
#define USE_TDM 1
#endif
#endif
#ifndef USE_TDM
#define USE_TDM 0
#endif

// Native bf16 convert (single v_cvt on gfx1250) instead of software RNE.
static __device__ __forceinline__ unsigned short f2bf(float f) {
  __bf16 h = (__bf16)f;
  return __builtin_bit_cast(unsigned short, h);
}
static __device__ __forceinline__ float bf2f(unsigned short h) {
  return (float)__builtin_bit_cast(__bf16, h);
}
// K-offset of element-pair p (p=0..3 -> K[0..15] half, p=4..7 -> K[16..31] half)
static __device__ __forceinline__ int kpair(int hi, int p) {
  return (p < 4) ? (hi * 8 + 2 * p) : (16 + hi * 8 + 2 * (p - 4));
}
static __device__ __forceinline__ v8f wmma_bf16(u32x8 a, u32x8 b, v8f c) {
  return __builtin_amdgcn_wmma_f32_16x16x32_bf16(
      false, __builtin_bit_cast(v16bf, a),
      false, __builtin_bit_cast(v16bf, b),
      (short)0, c, false, false);
}
// Load a 16x32 fragment from a row-major bf16 matrix (rows = M for A-frag,
// rows = N for B-frag since B is stored (n,k) row-major): pairs contiguous.
static __device__ __forceinline__ u32x8 load_frag(const unsigned short* base, int ld, int lane) {
  int m = lane & 15, hi = lane >> 4;
  const unsigned short* r = base + (size_t)m * ld;
  u32x8 f;
#pragma unroll
  for (int p = 0; p < 8; ++p)
    f[p] = *(const unsigned int*)(r + kpair(hi, p));
  return f;
}

// ---------------------------------------------------------------------------
// K0a: generic f32 -> bf16 convert
__global__ void k_cvt_bf16(const float* __restrict__ src, unsigned short* __restrict__ dst, int n) {
  int i = blockIdx.x * blockDim.x + threadIdx.x;
  if (i < n) dst[i] = f2bf(src[i]);
}

// K0b: pack W_q/W_k/W_v/W_alpha into one (256 x 1024) bf16 matrix (n,k) row-major
__global__ void k_pack_qkva(const float* __restrict__ Wq, const float* __restrict__ Wk,
                            const float* __restrict__ Wv, const float* __restrict__ Wa,
                            unsigned short* __restrict__ dst) {
  int i = blockIdx.x * blockDim.x + threadIdx.x;   // 256*1024 total
  int row = i >> 10, col = i & 1023;
  const float* s = (row < 64) ? Wq : (row < 128) ? Wk : (row < 192) ? Wv : Wa;
  dst[i] = f2bf(s[(size_t)(row & 63) * DM_ + col]);
}

// ---------------------------------------------------------------------------
// K1: G[j*128+c][d] = sum_v W_down[c][j*64+v] * W_up[j*64+v][d], bf16 out.
// One block per j (64 blocks, 8 waves). C = A(128x64) @ B(64x128).
__global__ void k1_buildG(const float* __restrict__ W_down, const float* __restrict__ W_up,
                          unsigned short* __restrict__ g_bf) {
  int j = blockIdx.x;
  int tid = threadIdx.x, lane = tid & 31, wave = tid >> 5;
  int m = lane & 15, hi = lane >> 4;
  int mt = wave;                       // M-tile (rows of C: c index)
  v8f acc[8] = {};
  for (int ks = 0; ks < 64; ks += 32) {
    u32x8 af;
#pragma unroll
    for (int p = 0; p < 8; ++p) {
      int k = ks + kpair(hi, p);
      const float* r = W_down + (size_t)(mt * 16 + m) * KV_ + j * 64 + k;
      af[p] = (unsigned)f2bf(r[0]) | ((unsigned)f2bf(r[1]) << 16);
    }
#pragma unroll
    for (int nt = 0; nt < 8; ++nt) {
      u32x8 bf_;
#pragma unroll
      for (int p = 0; p < 8; ++p) {
        int k = ks + kpair(hi, p);
        int col = nt * 16 + m;         // n = lane&15
        float b0 = W_up[(size_t)(j * 64 + k) * DC_ + col];
        float b1 = W_up[(size_t)(j * 64 + k + 1) * DC_ + col];
        bf_[p] = (unsigned)f2bf(b0) | ((unsigned)f2bf(b1) << 16);
      }
      acc[nt] = wmma_bf16(af, bf_, acc[nt]);
    }
  }
#pragma unroll
  for (int nt = 0; nt < 8; ++nt)
#pragma unroll
    for (int r = 0; r < 8; ++r) {
      int row = j * 128 + mt * 16 + r + 8 * hi;
      g_bf[(size_t)row * DC_ + nt * 16 + m] = f2bf(acc[nt][r]);
    }
}

// ---------------------------------------------------------------------------
// K2: fused projection: [q|k|v|alpha] = x(8192x1024) @ Wqkva^T + bias; sigmoid(alpha).
// Block tile: 64 tokens x 256 outputs; 8 waves; wave -> (mt = w&3, nt group = w>>2).
__global__ void k2_proj(const unsigned short* __restrict__ x_bf, const unsigned short* __restrict__ wqkva,
                        const float* __restrict__ b_q, const float* __restrict__ b_k,
                        const float* __restrict__ b_v, const float* __restrict__ b_a,
                        unsigned short* __restrict__ q_bf, unsigned short* __restrict__ k_bf,
                        unsigned short* __restrict__ v_bf, float* __restrict__ alpha_f) {
  int tid = threadIdx.x, lane = tid & 31, wave = tid >> 5;
  int hi = lane >> 4, n = lane & 15;
  int tokBase = blockIdx.x * 64;
  int mt = wave & 3, ntbase = (wave >> 2) * 8;
  v8f acc[8] = {};
  for (int ks = 0; ks < DM_; ks += 32) {
    // L2-resident weight stream: prefetch the next k-step's B tiles.
    if (ks + 32 < DM_)
      __builtin_prefetch(wqkva + (size_t)(ntbase * 16) * DM_ + ks + 32, 0, 1);
    u32x8 af = load_frag(x_bf + (size_t)(tokBase + mt * 16) * DM_ + ks, DM_, lane);
#pragma unroll
    for (int i = 0; i < 8; ++i) {
      u32x8 bf_ = load_frag(wqkva + (size_t)((ntbase + i) * 16) * DM_ + ks, DM_, lane);
      acc[i] = wmma_bf16(af, bf_, acc[i]);
    }
  }
#pragma unroll
  for (int i = 0; i < 8; ++i) {
    int ng = (ntbase + i) * 16 + n;
    int sel = ng >> 6, idx = ng & 63;
    const float* bias = (sel == 0) ? b_q : (sel == 1) ? b_k : (sel == 2) ? b_v : b_a;
    float bv = bias[idx];
#pragma unroll
    for (int r = 0; r < 8; ++r) {
      int token = tokBase + mt * 16 + r + 8 * hi;
      float val = acc[i][r] + bv;
      if (sel == 0)      q_bf[(size_t)token * DK_ + idx] = f2bf(val);
      else if (sel == 1) k_bf[(size_t)token * DK_ + idx] = f2bf(val);
      else if (sel == 2) v_bf[(size_t)token * DV_ + idx] = f2bf(val);
      else               alpha_f[(size_t)token * DK_ + idx] = 1.0f / (1.0f + __expf(-val));
    }
  }
}

// ---------------------------------------------------------------------------
// K3: b_in(8192x128) = KV(8192x4096) @ W_down^T; KV built on the fly (k ⊗ v).
// One block per 16 tokens; wave w owns N-tile w.
__global__ void k3_bin(const unsigned short* __restrict__ k_bf, const unsigned short* __restrict__ v_bf,
                       const unsigned short* __restrict__ wd_bf, float* __restrict__ b_in) {
  __shared__ unsigned short kt[16 * DK_];
  __shared__ unsigned short vt[16 * DV_];
  int tid = threadIdx.x, lane = tid & 31, wave = tid >> 5;
  int tok0 = blockIdx.x * 16;
  for (int i = tid; i < 16 * DK_; i += 256) {
    kt[i] = k_bf[(size_t)tok0 * DK_ + i];
    vt[i] = v_bf[(size_t)tok0 * DV_ + i];
  }
  __syncthreads();
  int m = lane & 15, hi = lane >> 4, nt = wave;
  v8f acc = {};
  for (int ks = 0; ks < KV_; ks += 32) {
    if (ks + 32 < KV_)
      __builtin_prefetch(wd_bf + (size_t)(nt * 16) * KV_ + ks + 32, 0, 1);
    u32x8 af;
#pragma unroll
    for (int p = 0; p < 8; ++p) {
      int kidx = ks + kpair(hi, p);
      int kk = kidx >> 6, vv = kidx & 63;
      float kf = bf2f(kt[m * DK_ + kk]);
      float v0 = bf2f(vt[m * DV_ + vv]);
      float v1 = bf2f(vt[m * DV_ + vv + 1]);
      af[p] = (unsigned)f2bf(kf * v0) | ((unsigned)f2bf(kf * v1) << 16);
    }
    u32x8 bf_ = load_frag(wd_bf + (size_t)(nt * 16) * KV_ + ks, KV_, lane);
    acc = wmma_bf16(af, bf_, acc);
  }
  int n = lane & 15;
#pragma unroll
  for (int r = 0; r < 8; ++r) {
    int mm = r + 8 * hi;
    b_in[(size_t)(tok0 + mm) * DC_ + nt * 16 + n] = acc[r];
  }
}

// ---------------------------------------------------------------------------
__global__ void k_init(unsigned int* arrive) {
  if (threadIdx.x < B_) arrive[threadIdx.x] = 0u;
}

// K4: the sequential scan. 64 blocks = batch(4) x slice(16). Slice g holds G rows
// [g*512, g*512+512) resident (TDM -> LDS -> VGPR A-fragments). Each step:
// 16 WMMAs (Gslice @ c), alpha-weighted j-reduction, cross-WG partial exchange
// through L2 with one agent-scope barrier per batch per step.
__global__ void k4_scan(const unsigned short* __restrict__ g_bf, const float* __restrict__ alpha_f,
                        const float* __restrict__ b_in, float* __restrict__ cs,
                        float* __restrict__ partials, unsigned int* __restrict__ arrive) {
  extern __shared__ unsigned char smem[];
  unsigned short* gt   = (unsigned short*)smem;                 // 512*128 bf16 = 128KB
  float* part_f        = (float*)(smem + 512 * DC_ * 2);        // 128 f32
  float* alpha_s       = part_f + DC_;                          // 4 (pad 8)
  unsigned short* c_bf = (unsigned short*)(alpha_s + 8);        // 128 bf16

  int tid = threadIdx.x, lane = tid & 31, wave = tid >> 5;
  int hi = lane >> 4, n = lane & 15;
  int b = blockIdx.x >> 4, g = blockIdx.x & 15;
  const unsigned short* gsrc = g_bf + (size_t)g * 512 * DC_;

#if USE_TDM
  if (tid < 32) {
    unsigned lds_off = (unsigned)(uintptr_t)gt;
    unsigned long long ga = (unsigned long long)(uintptr_t)gsrc;
    // D# group0: count=1 | lds_addr | global_addr[56:0] | type=2
    u32x4 g0 = { 1u, lds_off, (unsigned)(ga & 0xFFFFFFFFu),
                 (unsigned)((ga >> 32) & 0x01FFFFFFu) | 0x80000000u };
    // D# group1: data_size=2B; tensor_dim0=128, tensor_dim1=8192;
    // tile_dim0=128, tile_dim1=512; tensor_dim0_stride=128
    i32x8 g1 = { (int)0x00010000, (int)(128u << 16), (int)(8192u << 16),
                 (int)(128u << 16), 512, 128, 0, 0 };
    i32x4 z4 = { 0, 0, 0, 0 };
    i32x8 z8 = { 0, 0, 0, 0, 0, 0, 0, 0 };
    __builtin_amdgcn_tensor_load_to_lds(g0, g1, z4, z4, z8, 0);
    __builtin_amdgcn_s_wait_tensorcnt(0);
  }
#else
  for (int i = tid; i < 512 * DC_ / 2; i += 256)
    ((unsigned int*)gt)[i] = ((const unsigned int*)gsrc)[i];
#endif
  __syncthreads();

  // Pin all A-fragments of this slice in VGPRs (16 frags x 8 VGPRs).
  u32x8 afr[16];
#pragma unroll
  for (int i = 0; i < 4; ++i)
#pragma unroll
    for (int kk = 0; kk < 4; ++kk)
      afr[i * 4 + kk] = load_frag(gt + (size_t)((wave * 4 + i) * 16) * DC_ + kk * 32, DC_, lane);

  if (tid < DC_) c_bf[tid] = 0;   // c_{-1} = 0
  __syncthreads();

  for (int t = 0; t < S_; ++t) {
    if (tid < 4) alpha_s[tid] = alpha_f[(size_t)(b * S_ + t) * DK_ + g * 4 + tid];
    if (tid < DC_) part_f[tid] = 0.0f;
    __syncthreads();

    // B fragment: column 0 = c_{t-1}, rest zero.
    u32x8 bfr[4];
#pragma unroll
    for (int kk = 0; kk < 4; ++kk) {
      u32x8 f;
#pragma unroll
      for (int p = 0; p < 8; ++p) {
        unsigned vp = 0u;
        if (n == 0) vp = *(const unsigned int*)(c_bf + kk * 32 + kpair(hi, p));
        f[p] = vp;
      }
      bfr[kk] = f;
    }
#pragma unroll
    for (int i = 0; i < 4; ++i) {
      v8f acc = {};
#pragma unroll
      for (int kk = 0; kk < 4; ++kk)
        acc = wmma_bf16(afr[i * 4 + kk], bfr[kk], acc);
      if (n == 0) {
#pragma unroll
        for (int r = 0; r < 8; ++r) {
          int mrow = (wave * 4 + i) * 16 + r + 8 * hi;   // row within 512-slice
          int jl = mrow >> 7, c = mrow & 127;
          __hip_atomic_fetch_add(&part_f[c], alpha_s[jl] * acc[r],
                                 __ATOMIC_RELAXED, __HIP_MEMORY_SCOPE_WORKGROUP);
        }
      }
    }
    __syncthreads();

    float* pdst = partials + (((size_t)(t & 1) * B_ + b) * 16 + g) * DC_;
    if (tid < DC_) pdst[tid] = part_f[tid];
    __syncthreads();

    if (tid == 0) {
      __threadfence();
      __hip_atomic_fetch_add(arrive + b, 1u, __ATOMIC_ACQ_REL, __HIP_MEMORY_SCOPE_AGENT);
      unsigned target = 16u * (unsigned)(t + 1);
      while (__hip_atomic_load(arrive + b, __ATOMIC_ACQUIRE, __HIP_MEMORY_SCOPE_AGENT) < target)
        __builtin_amdgcn_s_sleep(1);
    }
    __syncthreads();

    if (tid < DC_) {
      const float* psrc = partials + ((size_t)(t & 1) * B_ + b) * 16 * DC_;
      float sum = b_in[(size_t)(b * S_ + t) * DC_ + tid];
#pragma unroll
      for (int gg = 0; gg < 16; ++gg)
        sum += __hip_atomic_load(psrc + gg * DC_ + tid, __ATOMIC_RELAXED, __HIP_MEMORY_SCOPE_AGENT);
      c_bf[tid] = f2bf(sum);
      if (g == 0) cs[(size_t)(b * S_ + t) * DC_ + tid] = sum;
    }
    __syncthreads();
  }
}

// ---------------------------------------------------------------------------
// K5: readout. Per 16-token tile: U(4096x16) = W_up @ cs_tile^T via WMMA,
// then o[s][v] += q[s][kk] * U[kk*64+v][s] scattered through ds_add_f32.
__global__ void k5_readout(const unsigned short* __restrict__ wu_bf, const float* __restrict__ cs,
                           const unsigned short* __restrict__ q_bf, float* __restrict__ out) {
  __shared__ unsigned short ct[16 * DC_];
  __shared__ unsigned short qt[16 * DK_];
  __shared__ float oacc[16 * DV_];
  int tid = threadIdx.x, lane = tid & 31, wave = tid >> 5;
  int tok0 = blockIdx.x * 16;
  for (int i = tid; i < 16 * DC_; i += 256) ct[i] = f2bf(cs[(size_t)tok0 * DC_ + i]);
  for (int i = tid; i < 16 * DK_; i += 256) { qt[i] = q_bf[(size_t)tok0 * DK_ + i]; oacc[i] = 0.0f; }
  __syncthreads();
  int hi = lane >> 4, n = lane & 15;
  u32x8 bfr[4];
#pragma unroll
  for (int kk = 0; kk < 4; ++kk)
    bfr[kk] = load_frag(ct + kk * 32, DC_, lane);   // B[k=c][n=s] from (s,c) row-major
  for (int it = 0; it < 32; ++it) {
    int mt = wave * 32 + it;                        // 256 M-tiles of W_up rows
    if (it + 1 < 32)
      __builtin_prefetch(wu_bf + (size_t)((mt + 1) * 16) * DC_, 0, 1);
    v8f acc = {};
#pragma unroll
    for (int kk = 0; kk < 4; ++kk) {
      u32x8 af = load_frag(wu_bf + (size_t)(mt * 16) * DC_ + kk * 32, DC_, lane);
      acc = wmma_bf16(af, bfr[kk], acc);
    }
#pragma unroll
    for (int r = 0; r < 8; ++r) {
      int m = mt * 16 + r + 8 * hi;                 // kv index
      int kh = m >> 6, v = m & 63;
      float val = bf2f(qt[n * DK_ + kh]) * acc[r];
      __hip_atomic_fetch_add(&oacc[n * DV_ + v], val,
                             __ATOMIC_RELAXED, __HIP_MEMORY_SCOPE_WORKGROUP);
    }
  }
  __syncthreads();
  for (int i = tid; i < 16 * DV_; i += 256) out[(size_t)tok0 * DV_ + i] = oacc[i];
}

// ---------------------------------------------------------------------------
extern "C" void kernel_launch(void* const* d_in, const int* in_sizes, int n_in,
                              void* d_out, int out_size, void* d_ws, size_t ws_size,
                              hipStream_t stream) {
  const float* x    = (const float*)d_in[0];
  const float* W_q  = (const float*)d_in[1];
  const float* b_q  = (const float*)d_in[2];
  const float* W_k  = (const float*)d_in[3];
  const float* b_k  = (const float*)d_in[4];
  const float* W_v  = (const float*)d_in[5];
  const float* b_v  = (const float*)d_in[6];
  const float* W_a  = (const float*)d_in[7];
  const float* b_a  = (const float*)d_in[8];
  const float* W_dn = (const float*)d_in[9];
  const float* W_up = (const float*)d_in[10];
  float* out = (float*)d_out;

  unsigned char* ws = (unsigned char*)d_ws;
  size_t off = 0;
  auto carve = [&](size_t bytes) { size_t o = off; off = (off + bytes + 255) & ~(size_t)255; return o; };
  unsigned short* x_bf   = (unsigned short*)(ws + carve((size_t)NTOK * DM_ * 2));   // 16 MB
  unsigned short* wqkva  = (unsigned short*)(ws + carve((size_t)256 * DM_ * 2));    // 512 KB
  unsigned short* wd_bf  = (unsigned short*)(ws + carve((size_t)DC_ * KV_ * 2));    // 1 MB
  unsigned short* wu_bf  = (unsigned short*)(ws + carve((size_t)KV_ * DC_ * 2));    // 1 MB
  unsigned short* g_bf   = (unsigned short*)(ws + carve((size_t)DK_ * DC_ * DC_ * 2)); // 2 MB
  unsigned short* q_bf   = (unsigned short*)(ws + carve((size_t)NTOK * DK_ * 2));
  unsigned short* k_bf   = (unsigned short*)(ws + carve((size_t)NTOK * DK_ * 2));
  unsigned short* v_bf   = (unsigned short*)(ws + carve((size_t)NTOK * DV_ * 2));
  float*          alphaf = (float*)(ws + carve((size_t)NTOK * DK_ * 4));            // 2 MB
  float*          b_in   = (float*)(ws + carve((size_t)NTOK * DC_ * 4));            // 4 MB
  float*          cs     = (float*)(ws + carve((size_t)NTOK * DC_ * 4));            // 4 MB
  float*          parts  = (float*)(ws + carve((size_t)2 * B_ * 16 * DC_ * 4));     // 64 KB
  unsigned int*   arrive = (unsigned int*)(ws + carve(256));
  if (off > ws_size) return;

  // Conversions / packing
  k_cvt_bf16<<<(NTOK * DM_) / 256, 256, 0, stream>>>(x, x_bf, NTOK * DM_);
  k_pack_qkva<<<(256 * DM_) / 256, 256, 0, stream>>>(W_q, W_k, W_v, W_a, wqkva);
  k_cvt_bf16<<<(DC_ * KV_) / 256, 256, 0, stream>>>(W_dn, wd_bf, DC_ * KV_);
  k_cvt_bf16<<<(KV_ * DC_) / 256, 256, 0, stream>>>(W_up, wu_bf, KV_ * DC_);

  // G = stack_j(M_j)
  k1_buildG<<<DK_, 256, 0, stream>>>(W_dn, W_up, g_bf);
  // Projections + sigmoid
  k2_proj<<<NTOK / 64, 256, 0, stream>>>(x_bf, wqkva, b_q, b_k, b_v, b_a,
                                         q_bf, k_bf, v_bf, alphaf);
  // b_in
  k3_bin<<<NTOK / 16, 256, 0, stream>>>(k_bf, v_bf, wd_bf, b_in);

  // Scan (persistent, 64 co-resident WGs, 132 KB dynamic LDS each)
  k_init<<<1, 64, 0, stream>>>(arrive);
  size_t smemK4 = (size_t)512 * DC_ * 2 + DC_ * 4 + 8 * 4 + DC_ * 2;
  k4_scan<<<B_ * 16, 256, smemK4, stream>>>(g_bf, alphaf, b_in, cs, parts, arrive);

  // Readout
  k5_readout<<<NTOK / 16, 256, 0, stream>>>(wu_bf, cs, q_bf, out);
}
